// SupConLoss_new_25417616458229
// MI455X (gfx1250) — matrix operand order repaired and checked by reading, not compiled, not run on hardware
//
#include <hip/hip_runtime.h>
#include <hip/hip_bf16.h>

// ---------------------------------------------------------------------------
// Fused supervised cross-entropy over normalized class centers, CDNA5/gfx1250.
//   c = center / ||center||_col ;  logits = features @ c ;  mean NLL(log_softmax)
// B=16384, D=512, C=1000 (padded to 1024 for the WMMA tiling).
// Precision: f16 inputs into V_WMMA_F32_16X16X32_F16, f32 accumulation.
// ---------------------------------------------------------------------------

typedef _Float16 v16h __attribute__((ext_vector_type(16)));
typedef _Float16 v8h  __attribute__((ext_vector_type(8)));
typedef float    v8f  __attribute__((ext_vector_type(8)));

#define KD 512          // reduction depth (feature dim)
#define ROWS_PER_BLK 16 // logits rows per block (one WMMA M tile)
#define WAVES_PER_BLK 8 // 256 threads
#define COLS_PER_WAVE 128

// ---------------------------------------------------------------------------
// Kernel 1: normalize center columns, emit f16, transposed to [col][k] so that
// a WMMA B-fragment is one contiguous 32-byte load per lane. Pad cols -> 0.
// ---------------------------------------------------------------------------
__global__ __launch_bounds__(256)
void center_norm_kernel(const float* __restrict__ center,  // [KD][C] row-major
                        _Float16* __restrict__ cT,         // [Cpad][KD]
                        int C, int Cpad) {
    __shared__ float ss[4][64];
    const int tx  = threadIdx.x & 63;   // column within tile
    const int ty  = threadIdx.x >> 6;   // 0..3
    const int col = blockIdx.x * 64 + tx;
    const bool valid = (col < C);

    float acc = 0.f;
    for (int d = ty; d < KD; d += 4) {             // coalesced across tx
        float x = valid ? center[(size_t)d * C + col] : 0.f;
        acc += x * x;
    }
    ss[ty][tx] = acc;
    __syncthreads();
    float s   = ss[0][tx] + ss[1][tx] + ss[2][tx] + ss[3][tx];
    float nrm = fmaxf(sqrtf(s), 1e-12f);
    float scale = valid ? (1.f / nrm) : 0.f;

    for (int d = ty; d < KD; d += 4) {
        float x = valid ? center[(size_t)d * C + col] : 0.f;
        cT[(size_t)col * KD + d] = (_Float16)(x * scale);
    }
}

// ---------------------------------------------------------------------------
// Kernel 2: fused GEMM (WMMA f16) + log-softmax + NLL partial per block.
// Block = 256 threads = 8 waves. Block owns 16 rows; wave w owns cols
// [w*128, w*128+128) as 8 accumulator fragments (16x16 f32 each).
// ---------------------------------------------------------------------------
__global__ __launch_bounds__(256)
void fused_gemm_nll_kernel(const float* __restrict__ features,  // [B][KD]
                           const int*   __restrict__ labels,    // [B]
                           const _Float16* __restrict__ cT,     // [Cpad][KD]
                           float* __restrict__ partial,         // [B/16]
                           int C) {
    __shared__ _Float16 sA[ROWS_PER_BLK * KD];      // 16 KB f16 A panel
    __shared__ float    redmax[WAVES_PER_BLK][16];
    __shared__ float    redsum[WAVES_PER_BLK][16];
    __shared__ float    labz[16];
    __shared__ int      labl[16];

    const int tid = threadIdx.x;
    const int r0  = blockIdx.x * ROWS_PER_BLK;

    if (tid < 16) labl[tid] = labels[r0 + tid];

    // Stage A panel: f32 -> f16 into LDS, fully coalesced.
    for (int idx = tid; idx < ROWS_PER_BLK * KD; idx += 256) {
        int m = idx >> 9;          // / KD
        int k = idx & (KD - 1);
        sA[idx] = (_Float16)features[(size_t)(r0 + m) * KD + k];
    }
    __syncthreads();

    const int lane  = tid & 31;
    const int w     = tid >> 5;
    const int mgrp  = lane & 15;          // row (A) / col (B,C,D) within tile
    const int rbase = (lane >> 4) * 8;    // C/D rows: lanes 0-15 -> 0..7, hi -> 8..15

    v8f acc[8];
    const v8f vzero = {0.f, 0.f, 0.f, 0.f, 0.f, 0.f, 0.f, 0.f};
#pragma unroll
    for (int t = 0; t < 8; ++t) acc[t] = vzero;

    const int kbA = (lane >> 4) * 16;     // A frag: lane half i -> K + 16i .. 16i+15 (contiguous)
    const int kbBlane = (lane >> 4) * 16; // B frag: halves i -> K + 16i .. 16i+15 (contiguous)

    for (int k0 = 0; k0 < KD; k0 += 32) {
        const _Float16* ap = &sA[mgrp * KD + k0 + kbA];
        v8h alo = *(const v8h*)(ap);
        v8h ahi = *(const v8h*)(ap + 8);
        v16h a  = __builtin_shufflevector(alo, ahi,
                    0, 1, 2, 3, 4, 5, 6, 7, 8, 9, 10, 11, 12, 13, 14, 15);
        const int kbB = k0 + kbBlane;
#pragma unroll
        for (int t = 0; t < 8; ++t) {
            const int col = w * COLS_PER_WAVE + t * 16 + mgrp;
            const _Float16* bp = cT + (size_t)col * KD + kbB;
            if (k0 + 32 < KD)
                __builtin_prefetch((const char*)bp + 64, 0, 1); // next K-step
            v16h b = *(const v16h*)bp;   // 32-byte contiguous, 32B-aligned
            acc[t] = __builtin_amdgcn_wmma_f32_16x16x32_f16(
                /*neg_a=*/false, a, /*neg_b=*/false, b,
                /*c_mod=*/(short)0, acc[t],
                /*reuse_a=*/false, /*reuse_b=*/false);
        }
    }

    // ---- row max (mask out padded cols >= C) ----
    float pmax[8];
#pragma unroll
    for (int r = 0; r < 8; ++r) pmax[r] = -3.0e38f;
#pragma unroll
    for (int t = 0; t < 8; ++t) {
        const int col = w * COLS_PER_WAVE + t * 16 + mgrp;
        if (col < C) {
#pragma unroll
            for (int r = 0; r < 8; ++r) pmax[r] = fmaxf(pmax[r], acc[t][r]);
        }
    }
#pragma unroll
    for (int off = 8; off >= 1; off >>= 1)
#pragma unroll
        for (int r = 0; r < 8; ++r)
            pmax[r] = fmaxf(pmax[r], __shfl_xor(pmax[r], off, 32));
    if (mgrp == 0)
#pragma unroll
        for (int r = 0; r < 8; ++r) redmax[w][rbase + r] = pmax[r];

    // ---- extract label logit (exactly one owner lane per row) ----
#pragma unroll
    for (int t = 0; t < 8; ++t) {
        const int col = w * COLS_PER_WAVE + t * 16 + mgrp;
#pragma unroll
        for (int r = 0; r < 8; ++r)
            if (col == labl[rbase + r]) labz[rbase + r] = acc[t][r];
    }
    __syncthreads();

    // ---- global row max, then sum of exp ----
    float rowmax[8];
#pragma unroll
    for (int r = 0; r < 8; ++r) {
        float m = redmax[0][rbase + r];
#pragma unroll
        for (int ww = 1; ww < WAVES_PER_BLK; ++ww)
            m = fmaxf(m, redmax[ww][rbase + r]);
        rowmax[r] = m;
    }
    float psum[8];
#pragma unroll
    for (int r = 0; r < 8; ++r) psum[r] = 0.f;
#pragma unroll
    for (int t = 0; t < 8; ++t) {
        const int col = w * COLS_PER_WAVE + t * 16 + mgrp;
        if (col < C) {
#pragma unroll
            for (int r = 0; r < 8; ++r)
                psum[r] += __expf(acc[t][r] - rowmax[r]);
        }
    }
#pragma unroll
    for (int off = 8; off >= 1; off >>= 1)
#pragma unroll
        for (int r = 0; r < 8; ++r)
            psum[r] += __shfl_xor(psum[r], off, 32);
    if (mgrp == 0)
#pragma unroll
        for (int r = 0; r < 8; ++r) redsum[w][rbase + r] = psum[r];
    __syncthreads();

    // ---- per-block NLL partial sum ----
    if (tid == 0) {
        float tot = 0.f;
        for (int row = 0; row < 16; ++row) {
            float m = redmax[0][row];
            float s = redsum[0][row];
            for (int ww = 1; ww < WAVES_PER_BLK; ++ww) {
                m = fmaxf(m, redmax[ww][row]);
                s += redsum[ww][row];
            }
            tot += -(labz[row] - m - __logf(s));
        }
        partial[blockIdx.x] = tot;
    }
}

// ---------------------------------------------------------------------------
// Kernel 3: deterministic tree reduction of per-block partials -> mean.
// ---------------------------------------------------------------------------
__global__ __launch_bounds__(256)
void reduce_partials_kernel(const float* __restrict__ partial, int n,
                            float invB, float* __restrict__ out) {
    __shared__ float sm[256];
    float s = 0.f;
    for (int i = threadIdx.x; i < n; i += 256) s += partial[i];
    sm[threadIdx.x] = s;
    __syncthreads();
    for (int off = 128; off > 0; off >>= 1) {
        if (threadIdx.x < off) sm[threadIdx.x] += sm[threadIdx.x + off];
        __syncthreads();
    }
    if (threadIdx.x == 0) out[0] = sm[0] * invB;
}

// ---------------------------------------------------------------------------
extern "C" void kernel_launch(void* const* d_in, const int* in_sizes, int n_in,
                              void* d_out, int out_size, void* d_ws, size_t ws_size,
                              hipStream_t stream) {
    const float* features = (const float*)d_in[0];   // [B][512] f32
    const int*   labels   = (const int*)d_in[1];     // [B] int32
    const float* center   = (const float*)d_in[2];   // [512][C] f32

    const int B    = in_sizes[1];                    // 16384
    const int C    = in_sizes[2] / KD;               // 1000
    const int Cpad = ((C + WAVES_PER_BLK * COLS_PER_WAVE - 1)
                      / (WAVES_PER_BLK * COLS_PER_WAVE))
                      * (WAVES_PER_BLK * COLS_PER_WAVE);  // 1024
    const int nBlocks = B / ROWS_PER_BLK;            // 1024

    // Workspace layout: [cT f16: Cpad*KD] [partials f32: nBlocks]
    _Float16* cT      = (_Float16*)d_ws;
    float*    partial = (float*)((char*)d_ws + (size_t)Cpad * KD * sizeof(_Float16));

    center_norm_kernel<<<Cpad / 64, 256, 0, stream>>>(center, cT, C, Cpad);
    fused_gemm_nll_kernel<<<nBlocks, 256, 0, stream>>>(features, labels, cT,
                                                       partial, C);
    reduce_partials_kernel<<<1, 256, 0, stream>>>(partial, nBlocks,
                                                  1.0f / (float)B, d_out ? (float*)d_out : nullptr);
    (void)n_in; (void)out_size; (void)ws_size;
}
